// MultiHeadAttention_86998857548450
// MI455X (gfx1250) — matrix operand order repaired and checked by reading, not compiled, not run on hardware
//
#include <hip/hip_runtime.h>

typedef __attribute__((ext_vector_type(16))) __bf16 v16bf;
typedef __attribute__((ext_vector_type(8)))  __bf16 v8bf;
typedef __attribute__((ext_vector_type(8)))  float  v8f;
typedef __attribute__((ext_vector_type(4)))  float  v4f;
typedef __attribute__((ext_vector_type(4)))  unsigned int u32x4;
typedef __attribute__((ext_vector_type(8)))  unsigned int u32x8;

#define B_ 4
#define S_ 2048
#define D_ 1024
#define H_ 16
#define E_ 64            // DQKV
#define M_ (B_ * S_)     // 8192 rows of activations
#define KP_ 256          // GEMM k-panel staged to LDS by the TDM

// ---------------------------------------------------------------------------
// WMMA fragment loaders (wave32, per CDNA5 ISA §7.12.2)
// A (16x32 bf16, MxK): lane m=lane&15, kh=8*(lane>>4); elems 0..7 = K[kh..kh+7],
//                      elems 8..15 = K[kh+16..kh+23]  -> two 16B loads.
// B (32x16 bf16, KxN): lane n=lane&15, kb=8*(lane>>4); elems 0..7 = K[kb..kb+7],
//                      elems 8..15 = K[kb+16..kb+23] for column n -> two 16B loads.
// C/D (16x16 f32):     vgpr r <-> row r+8*(lane>>4), lane&15 <-> col.
// ---------------------------------------------------------------------------
__device__ __forceinline__ v16bf load_a_frag_bf16(const __bf16* __restrict__ base,
                                                  int ld, int lane) {
  const int m  = lane & 15;
  const int kh = (lane >> 4) << 3;
  const __bf16* p = base + (size_t)m * ld + kh;
  union { v16bf v; v8bf h[2]; } u;
  u.h[0] = *reinterpret_cast<const v8bf*>(p);
  u.h[1] = *reinterpret_cast<const v8bf*>(p + 16);
  return u.v;
}

// Same A layout but sourced from fp32 (on-the-fly bf16 conversion).
__device__ __forceinline__ v16bf load_a_frag_f32(const float* __restrict__ base,
                                                 int ld, int lane) {
  const int m  = lane & 15;
  const int kh = (lane >> 4) << 3;
  const float* p = base + (size_t)m * ld + kh;
  v4f f0 = *reinterpret_cast<const v4f*>(p);
  v4f f1 = *reinterpret_cast<const v4f*>(p + 4);
  v4f f2 = *reinterpret_cast<const v4f*>(p + 16);
  v4f f3 = *reinterpret_cast<const v4f*>(p + 20);
  union { v16bf v; __bf16 e[16]; } u;
#pragma unroll
  for (int i = 0; i < 4; ++i) {
    u.e[i]      = (__bf16)f0[i];
    u.e[4 + i]  = (__bf16)f1[i];
    u.e[8 + i]  = (__bf16)f2[i];
    u.e[12 + i] = (__bf16)f3[i];
  }
  return u.v;
}

// B fragment: source must have column n's K-values contiguous at base + n*ld.
__device__ __forceinline__ v16bf load_b_frag(const __bf16* __restrict__ base,
                                             int ld, int lane) {
  const int n  = lane & 15;
  const int kb = (lane >> 4) << 3;
  const __bf16* p = base + (size_t)n * ld + kb;
  union { v16bf v; v8bf h[2]; } u;
  u.h[0] = *reinterpret_cast<const v8bf*>(p);
  u.h[1] = *reinterpret_cast<const v8bf*>(p + 16);
  return u.v;
}

__device__ __forceinline__ v8f wmma_bf16(v16bf a, v16bf b, v8f c) {
  // (neg_a, A, neg_b, B, c_mod, C, reuse_a, reuse_b)
  return __builtin_amdgcn_wmma_f32_16x16x32_bf16(false, a, false, b, (short)0, c,
                                                 false, false);
}

// ---------------------------------------------------------------------------
// Tensor Data Mover: DMA a 2D bf16 tile (tile_k x tile_rows, row stride in
// elements) from global memory into LDS. 2-group D# (CDNA5 ISA):
//   group0: count=1 | lds_addr | global_addr | type=2
//   group1: data_size=2B, tensor dims (large, in-bounds), tile dims, stride0.
// Tracked with TENSORcnt; caller waits s_wait_tensorcnt then barriers.
// ---------------------------------------------------------------------------
__device__ __forceinline__ void tdm_load_2d_bf16(unsigned lds_off, const void* gaddr,
                                                 unsigned tile_k, unsigned tile_rows,
                                                 unsigned row_stride_elems) {
  const unsigned long long ga = (unsigned long long)gaddr;
  u32x4 g0;
  g0[0] = 1u;                                          // count=1, user mode
  g0[1] = lds_off;                                     // lds_addr (bytes)
  g0[2] = (unsigned)ga;                                // global_addr[31:0]
  g0[3] = (unsigned)((ga >> 32) & 0x01FFFFFFu)         // global_addr[56:32]
        | (2u << 30);                                  // type=2 ("image")
  u32x8 g1;
  g1[0] = 1u << 16;                                    // data_size=1 -> 2 bytes
  g1[1] = 0u;                                          // tensor_dim0 = 1<<20 (lsb16=0)
  g1[2] = 0x10u;                                       // dim0 msb | dim1 lsb16=0
  g1[3] = 0x10u | (tile_k << 16);                      // dim1 msb | tile_dim0
  g1[4] = tile_rows;                                   // tile_dim1 | tile_dim2=0
  g1[5] = row_stride_elems;                            // tensor_dim0_stride[31:0]
  g1[6] = 0u;                                          // stride0 msb | stride1 lsb
  g1[7] = 0u;
  asm volatile("tensor_load_to_lds %0, %1" :: "s"(g0), "s"(g1) : "memory");
}

// ---------------------------------------------------------------------------
// Weight fp32 -> bf16 conversion (one-shot, tiny vs. GEMM cost)
// ---------------------------------------------------------------------------
__global__ void cvt_f32_to_bf16(const float* __restrict__ src,
                                __bf16* __restrict__ dst, int n) {
  int i = (blockIdx.x * blockDim.x + threadIdx.x) * 4;
  if (i + 3 < n) {
    v4f f = *reinterpret_cast<const v4f*>(src + i);
    dst[i + 0] = (__bf16)f[0];
    dst[i + 1] = (__bf16)f[1];
    dst[i + 2] = (__bf16)f[2];
    dst[i + 3] = (__bf16)f[3];
  } else {
    for (; i < n; ++i) dst[i] = (__bf16)src[i];
  }
}

// ---------------------------------------------------------------------------
// QKV projection GEMM: C[m,n] = sum_k X[m,k] * W[n,k].
// Block: 256 thr = 8 waves; tile 256(M) x 64(N); wave strip = 32(M) x 64(N).
// W panel (64 x KP_ bf16 = 32 KB) is TDM-DMA'd to LDS, double buffered; the
// next panel's DMA overlaps the current panel's 64 WMMAs per wave.
//   transposed=0: out [B,H,S,E] bf16 (Q,K);  transposed=1: out [B,H,E,S] (V).
// outScale folds the 1/sqrt(E) attention scale into Q.
// ---------------------------------------------------------------------------
__global__ void __launch_bounds__(256)
qkv_proj_gemm(const float* __restrict__ X, const __bf16* __restrict__ Wb,
              const float* __restrict__ bias, __bf16* __restrict__ out,
              float outScale, int transposed) {
  __shared__ __bf16 Bpanel[2][64 * KP_];   // 2 x 32 KB
  const int lane  = threadIdx.x & 31;
  const int wave  = threadIdx.x >> 5;
  const int mtile = blockIdx.x * 256 + wave * 32;
  const int nbase = blockIdx.y * 64;
  const __bf16* Wpan = Wb + (size_t)nbase * D_;

  if (wave == 0) {
    tdm_load_2d_bf16((unsigned)(size_t)&Bpanel[0][0], Wpan, KP_, 64, D_);
    __builtin_amdgcn_s_wait_tensorcnt(0);
  }
  __syncthreads();

  v8f acc[2][4] = { { {}, {}, {}, {} }, { {}, {}, {}, {} } };
  const float* A0 = X + (size_t)mtile * D_;
  const float* A1 = A0 + (size_t)16 * D_;

  for (int p = 0; p < D_ / KP_; ++p) {
    if (p + 1 < D_ / KP_ && wave == 0)
      tdm_load_2d_bf16((unsigned)(size_t)&Bpanel[(p + 1) & 1][0],
                       Wpan + (size_t)(p + 1) * KP_, KP_, 64, D_);
    const __bf16* Bp = &Bpanel[p & 1][0];
#pragma unroll
    for (int kk = 0; kk < KP_; kk += 32) {
      const int k0 = p * KP_ + kk;
      v16bf a0 = load_a_frag_f32(A0 + k0, D_, lane);
      v16bf a1 = load_a_frag_f32(A1 + k0, D_, lane);
#pragma unroll
      for (int t = 0; t < 4; ++t) {
        v16bf b = load_b_frag(Bp + (size_t)t * 16 * KP_ + kk, KP_, lane);
        acc[0][t] = wmma_bf16(a0, b, acc[0][t]);
        acc[1][t] = wmma_bf16(a1, b, acc[1][t]);
      }
    }
    if (p + 1 < D_ / KP_) {
      if (wave == 0) __builtin_amdgcn_s_wait_tensorcnt(0);
      __syncthreads();
    }
  }

  const int lcol = lane & 15, lhalf = lane >> 4;
#pragma unroll
  for (int msub = 0; msub < 2; ++msub) {
#pragma unroll
    for (int t = 0; t < 4; ++t) {
      const int n  = nbase + t * 16 + lcol;
      const float bs = bias[n];
      const int h = n >> 6, e = n & (E_ - 1);
#pragma unroll
      for (int r = 0; r < 8; ++r) {
        const int m = mtile + msub * 16 + r + (lhalf << 3);
        const int b = m >> 11, s = m & (S_ - 1);
        const float v = (acc[msub][t][r] + bs) * outScale;
        const size_t idx = transposed
            ? ((size_t)(b * H_ + h) * E_ + e) * S_ + s
            : ((size_t)(b * H_ + h) * S_ + s) * E_ + e;
        out[idx] = (__bf16)v;
      }
    }
  }
}

// ---------------------------------------------------------------------------
// Flash attention: grid (S/64, B*H), block 128 = 4 waves.
// Each wave owns a 16-row q-tile; streams keys in chunks of 32 with online
// softmax; P transposed C-layout -> A-layout through a wave-private LDS tile.
// Q is pre-scaled by 1/8, so scores = QK^T directly.
// ---------------------------------------------------------------------------
__global__ void __launch_bounds__(128)
flash_attn_kernel(const __bf16* __restrict__ Q, const __bf16* __restrict__ K,
                  const __bf16* __restrict__ Vt, __bf16* __restrict__ ctx) {
  __shared__ __bf16 Pbuf[4][16 * 32];   // 1KB per wave

  const int lane = threadIdx.x & 31;
  const int wave = threadIdx.x >> 5;
  const int bh   = blockIdx.y;
  const int qrow = blockIdx.x * 64 + wave * 16;
  const int lcol = lane & 15, lhalf = lane >> 4;

  const __bf16* Qp = Q  + ((size_t)bh * S_ + qrow) * E_;
  const __bf16* Kp = K  + (size_t)bh * S_ * E_;
  const __bf16* Vp = Vt + (size_t)bh * E_ * S_;

  const v16bf aq0 = load_a_frag_bf16(Qp, E_, lane);        // dims 0..31
  const v16bf aq1 = load_a_frag_bf16(Qp + 32, E_, lane);   // dims 32..63

  float mrow[8], lrow[8];
#pragma unroll
  for (int r = 0; r < 8; ++r) { mrow[r] = -3.0e38f; lrow[r] = 0.0f; }
  v8f o0 = {}, o1 = {}, o2 = {}, o3 = {};
  __bf16* Pw = &Pbuf[wave][0];

  for (int kv = 0; kv < S_; kv += 32) {
    if (kv + 32 < S_) {
      __builtin_prefetch(Kp + (size_t)(kv + 32) * E_, 0, 0);
      __builtin_prefetch(Vp + (kv + 32), 0, 0);
    }
    // ---- scores: two 16x16 tiles over 32 keys (K rows are B columns) ----
    v8f s0 = {}, s1 = {};
    s0 = wmma_bf16(aq0, load_b_frag(Kp + (size_t)kv * E_,             E_, lane), s0);
    s0 = wmma_bf16(aq1, load_b_frag(Kp + (size_t)kv * E_ + 32,        E_, lane), s0);
    s1 = wmma_bf16(aq0, load_b_frag(Kp + (size_t)(kv + 16) * E_,      E_, lane), s1);
    s1 = wmma_bf16(aq1, load_b_frag(Kp + (size_t)(kv + 16) * E_ + 32, E_, lane), s1);

    // ---- online softmax; row stats via 16-lane butterflies ----
#pragma unroll
    for (int r = 0; r < 8; ++r) {
      float tmax = fmaxf(s0[r], s1[r]);
#pragma unroll
      for (int msk = 1; msk < 16; msk <<= 1)
        tmax = fmaxf(tmax, __shfl_xor(tmax, msk, 32));
      const float nm    = fmaxf(mrow[r], tmax);
      const float alpha = __expf(mrow[r] - nm);
      const float p0    = __expf(s0[r] - nm);
      const float p1    = __expf(s1[r] - nm);
      float psum = p0 + p1;
#pragma unroll
      for (int msk = 1; msk < 16; msk <<= 1)
        psum += __shfl_xor(psum, msk, 32);
      lrow[r] = lrow[r] * alpha + psum;
      mrow[r] = nm;
      o0[r] *= alpha; o1[r] *= alpha; o2[r] *= alpha; o3[r] *= alpha;
      const int m = r + (lhalf << 3);                 // C-layout row
      Pw[m * 32 + lcol]      = (__bf16)p0;
      Pw[m * 32 + 16 + lcol] = (__bf16)p1;
    }
    // DS ops are in-order per wave; fence to stop compiler reordering.
    asm volatile("s_wait_dscnt 0" ::: "memory");

    // ---- PV: P(16x32) as A-fragment from LDS, V dim-major as B ----
    const v16bf pa = load_a_frag_bf16(Pw, 32, lane);
    o0 = wmma_bf16(pa, load_b_frag(Vp + kv,           S_, lane), o0);
    o1 = wmma_bf16(pa, load_b_frag(Vp + 16 * S_ + kv, S_, lane), o1);
    o2 = wmma_bf16(pa, load_b_frag(Vp + 32 * S_ + kv, S_, lane), o2);
    o3 = wmma_bf16(pa, load_b_frag(Vp + 48 * S_ + kv, S_, lane), o3);
  }

  // ---- normalize + store ctx as [B, S, H*E] bf16 (A-matrix for out proj) ----
  const int b = bh >> 4, h = bh & (H_ - 1);
  __bf16* cp = ctx + (size_t)b * S_ * D_;
#pragma unroll
  for (int r = 0; r < 8; ++r) {
    const float inv = 1.0f / lrow[r];
    const int s = qrow + r + (lhalf << 3);
    const size_t rowoff = (size_t)s * D_ + h * E_ + lcol;
    cp[rowoff +  0] = (__bf16)(o0[r] * inv);
    cp[rowoff + 16] = (__bf16)(o1[r] * inv);
    cp[rowoff + 32] = (__bf16)(o2[r] * inv);
    cp[rowoff + 48] = (__bf16)(o3[r] * inv);
  }
}

// ---------------------------------------------------------------------------
// Output projection: out[m,d] = sum_f ctx[m,f] * wo[d,f] + bo[d], f32 out.
// Same TDM-staged, double-buffered structure as the QKV GEMM; A is bf16.
// ---------------------------------------------------------------------------
__global__ void __launch_bounds__(256)
out_proj_gemm(const __bf16* __restrict__ A, const __bf16* __restrict__ Wb,
              const float* __restrict__ bias, float* __restrict__ out) {
  __shared__ __bf16 Bpanel[2][64 * KP_];
  const int lane  = threadIdx.x & 31;
  const int wave  = threadIdx.x >> 5;
  const int mtile = blockIdx.x * 256 + wave * 32;
  const int nbase = blockIdx.y * 64;
  const __bf16* Wpan = Wb + (size_t)nbase * D_;

  if (wave == 0) {
    tdm_load_2d_bf16((unsigned)(size_t)&Bpanel[0][0], Wpan, KP_, 64, D_);
    __builtin_amdgcn_s_wait_tensorcnt(0);
  }
  __syncthreads();

  v8f acc[2][4] = { { {}, {}, {}, {} }, { {}, {}, {}, {} } };
  const __bf16* A0 = A + (size_t)mtile * D_;
  const __bf16* A1 = A0 + (size_t)16 * D_;

  for (int p = 0; p < D_ / KP_; ++p) {
    if (p + 1 < D_ / KP_ && wave == 0)
      tdm_load_2d_bf16((unsigned)(size_t)&Bpanel[(p + 1) & 1][0],
                       Wpan + (size_t)(p + 1) * KP_, KP_, 64, D_);
    const __bf16* Bp = &Bpanel[p & 1][0];
#pragma unroll
    for (int kk = 0; kk < KP_; kk += 32) {
      const int k0 = p * KP_ + kk;
      v16bf a0 = load_a_frag_bf16(A0 + k0, D_, lane);
      v16bf a1 = load_a_frag_bf16(A1 + k0, D_, lane);
#pragma unroll
      for (int t = 0; t < 4; ++t) {
        v16bf b = load_b_frag(Bp + (size_t)t * 16 * KP_ + kk, KP_, lane);
        acc[0][t] = wmma_bf16(a0, b, acc[0][t]);
        acc[1][t] = wmma_bf16(a1, b, acc[1][t]);
      }
    }
    if (p + 1 < D_ / KP_) {
      if (wave == 0) __builtin_amdgcn_s_wait_tensorcnt(0);
      __syncthreads();
    }
  }

  const int lcol = lane & 15, lhalf = lane >> 4;
#pragma unroll
  for (int msub = 0; msub < 2; ++msub) {
#pragma unroll
    for (int t = 0; t < 4; ++t) {
      const int n  = nbase + t * 16 + lcol;
      const float bs = bias[n];
#pragma unroll
      for (int r = 0; r < 8; ++r) {
        const int m = mtile + msub * 16 + r + (lhalf << 3);
        out[(size_t)m * D_ + n] = acc[msub][t][r] + bs;
      }
    }
  }
}

// ---------------------------------------------------------------------------
extern "C" void kernel_launch(void* const* d_in, const int* in_sizes, int n_in,
                              void* d_out, int out_size, void* d_ws, size_t ws_size,
                              hipStream_t stream) {
  const float* query = (const float*)d_in[0];
  const float* key   = (const float*)d_in[1];
  const float* value = (const float*)d_in[2];
  const float* wq    = (const float*)d_in[3];
  const float* bq    = (const float*)d_in[4];
  const float* wk    = (const float*)d_in[5];
  const float* bk    = (const float*)d_in[6];
  const float* wv    = (const float*)d_in[7];
  const float* bv    = (const float*)d_in[8];
  const float* wo    = (const float*)d_in[9];
  const float* bo    = (const float*)d_in[10];

  const size_t NW = (size_t)D_ * D_;   // 1M elems per weight matrix
  const size_t NX = (size_t)M_ * D_;   // 8M elems per activation tensor

  __bf16* Wqb = (__bf16*)d_ws;
  __bf16* Wkb = Wqb + NW;
  __bf16* Wvb = Wkb + NW;
  __bf16* Wob = Wvb + NW;
  __bf16* Qb  = Wob + NW;              // [B,H,S,E]   (pre-scaled by 1/8)
  __bf16* Kb  = Qb + NX;               // [B,H,S,E]
  __bf16* Vtb = Kb + NX;               // [B,H,E,S]   (dim-major)
  __bf16* Ctx = Vtb + NX;              // [B,S,H*E]
  // total ws: (4*NW + 4*NX) * 2 bytes = 72 MB

  const int cvtBlocks = (int)((NW / 4 + 255) / 256);
  cvt_f32_to_bf16<<<cvtBlocks, 256, 0, stream>>>(wq, Wqb, (int)NW);
  cvt_f32_to_bf16<<<cvtBlocks, 256, 0, stream>>>(wk, Wkb, (int)NW);
  cvt_f32_to_bf16<<<cvtBlocks, 256, 0, stream>>>(wv, Wvb, (int)NW);
  cvt_f32_to_bf16<<<cvtBlocks, 256, 0, stream>>>(wo, Wob, (int)NW);

  const dim3 gemmGrid(M_ / 256, D_ / 64);   // (32, 16)
  qkv_proj_gemm<<<gemmGrid, 256, 0, stream>>>(query, Wqb, bq, Qb, 0.125f, 0);
  qkv_proj_gemm<<<gemmGrid, 256, 0, stream>>>(key,   Wkb, bk, Kb,  1.0f,  0);
  qkv_proj_gemm<<<gemmGrid, 256, 0, stream>>>(value, Wvb, bv, Vtb, 1.0f,  1);

  flash_attn_kernel<<<dim3(S_ / 64, B_ * H_), 128, 0, stream>>>(Qb, Kb, Vtb, Ctx);

  out_proj_gemm<<<gemmGrid, 256, 0, stream>>>(Ctx, Wob, bo, (float*)d_out);
}
